// GCN_87875030876624
// MI455X (gfx1250) — matrix operand order repaired and checked by reading, not compiled, not run on hardware
//
#include <hip/hip_runtime.h>
#include <hip/hip_bf16.h>

// ---------------------------------------------------------------------------
// GCN (3x GCNConv) for MI455X / gfx1250.
//   y = GCNconv3( relu(GCNconv2( relu(GCNconv1(x)) )) )
// Dense lifts: V_WMMA_F32_16X16X32_BF16, one 16x16 tile per wave32, fully
// unrolled K loop, contiguous b128 fragment loads (weights pre-transposed to
// bf16 [Nout][K] so each lane's B fragment is 32 contiguous bytes).
// Aggregation: global f32 atomics; working set fits the 192MB L2.
// ---------------------------------------------------------------------------

typedef __attribute__((ext_vector_type(16))) __bf16 v16bf;
typedef __attribute__((ext_vector_type(8)))  float  v8f;

// ---------------- degree / normalization ----------------------------------

__global__ void k_init_deg(float* deg, int n) {
  int i = blockIdx.x * blockDim.x + threadIdx.x;
  if (i < n) deg[i] = 1.0f;                      // self-loop contribution
}

__global__ void k_count_deg(const long long* __restrict__ dst, float* deg, int e) {
  int i = blockIdx.x * blockDim.x + threadIdx.x;
  if (i < e) atomicAdd(&deg[(int)dst[i]], 1.0f);
}

__global__ void k_rsqrt_inplace(float* d, int n) {
  int i = blockIdx.x * blockDim.x + threadIdx.x;
  if (i < n) {
    float v = d[i];
    d[i] = (v > 0.0f) ? rsqrtf(v) : 0.0f;
  }
}

__global__ void k_edge_norm(const long long* __restrict__ src,
                            const long long* __restrict__ dst,
                            const float* __restrict__ dinv,
                            float* __restrict__ norm, int e) {
  int i = blockIdx.x * blockDim.x + threadIdx.x;
  if (i < e) norm[i] = dinv[(int)src[i]] * dinv[(int)dst[i]];
}

__global__ void k_zero(float* p, long n) {
  long i = (long)blockIdx.x * blockDim.x + threadIdx.x;
  if (i < n) p[i] = 0.0f;
}

// ---------------- weight transpose + bf16 quantize -------------------------
// Wt[n*K + k] = bf16( W[k*Nout + n] )  -> each lane's B fragment contiguous.

__global__ void k_weight_t_bf16(const float* __restrict__ W,
                                __bf16* __restrict__ Wt, int K, int Nout) {
  int i = blockIdx.x * blockDim.x + threadIdx.x;
  if (i >= K * Nout) return;
  int n = i / K, k = i % K;
  Wt[i] = (__bf16)W[k * Nout + n];
}

// ---------------- WMMA GEMM: Out[M,NOUT] = A[M,K] @ W[K,NOUT] --------------
// One wave32 per 16x16 tile; blockDim.x = 32*(NOUT/16); gridDim.x = M/16.
// Lane fragment layouts used here (A and B share one k order):
//   A 16x32 bf16 : lane (l<16 -> M=l, K in {0..15}; l>=16 -> +16)
//                  element e<8 : K = laneHalf*16 + e        (contiguous)
//                  element e>=8: K = laneHalf*16 + e (contiguous)
//   B 32x16 bf16 : lane N=l%16, element e: K = laneHalf*16 + e (contiguous)
//   C/D 16x16 f32: VGPR r -> M = r + laneHalf*8, N = l%16

template <int K, int NOUT>
__global__ void k_gemm_bf16_wmma(const float* __restrict__ A,
                                 const __bf16* __restrict__ Wt,  // [NOUT][K]
                                 float* __restrict__ Out) {
  const int wave     = threadIdx.x >> 5;
  const int lane     = threadIdx.x & 31;
  const int tileM    = blockIdx.x;
  const int tileN    = wave;
  const int laneHalf = lane >> 4;
  const int lmod     = lane & 15;

  const float*  arow = A  + (size_t)(tileM * 16 + lmod) * K;  // lane's A row
  const __bf16* wrow = Wt + (size_t)(tileN * 16 + lmod) * K;  // lane's B column

  v8f acc = {};
#pragma unroll
  for (int k0 = 0; k0 < K; k0 += 32) {
    const float* pa0 = arow + k0 + laneHalf * 16;      // K run 0..7 of half
    const float* pa1 = arow + k0 + laneHalf * 16 + 8;  // K run 8..15 of half
    float4 a0 = *(const float4*)(pa0);
    float4 a1 = *(const float4*)(pa0 + 4);
    float4 a2 = *(const float4*)(pa1);
    float4 a3 = *(const float4*)(pa1 + 4);
    v16bf a = {(__bf16)a0.x, (__bf16)a0.y, (__bf16)a0.z, (__bf16)a0.w,
               (__bf16)a1.x, (__bf16)a1.y, (__bf16)a1.z, (__bf16)a1.w,
               (__bf16)a2.x, (__bf16)a2.y, (__bf16)a2.z, (__bf16)a2.w,
               (__bf16)a3.x, (__bf16)a3.y, (__bf16)a3.z, (__bf16)a3.w};
    v16bf b = *(const v16bf*)(wrow + k0 + laneHalf * 16);  // 32B contiguous
    acc = __builtin_amdgcn_wmma_f32_16x16x32_bf16(
        /*neg_a=*/false, a, /*neg_b=*/false, b,
        /*c_mod=*/(short)0, acc, /*reuse_a=*/false, /*reuse_b=*/false);
  }

#pragma unroll
  for (int r = 0; r < 8; ++r) {
    const size_t m = (size_t)(tileM * 16 + r + laneHalf * 8);
    Out[m * (size_t)NOUT + (size_t)(tileN * 16 + lmod)] = acc[r];
  }
}

// ---------------- edge aggregation: agg[dst] += xw[src] * norm -------------

__global__ void k_aggregate(const long long* __restrict__ src,
                            const long long* __restrict__ dst,
                            const float* __restrict__ norm,
                            const float* __restrict__ xw,
                            float* __restrict__ agg, int E, int F) {
  int e = blockIdx.x * blockDim.x + threadIdx.x;
  if (e >= E) return;
  const int s = (int)src[e];
  const int d = (int)dst[e];
  const float w = norm[e];
  const float4* xs = (const float4*)(xw + (size_t)s * F);
  float* ad = agg + (size_t)d * F;
  const int nc = F >> 2;
  for (int c = 0; c < nc; ++c) {
    float4 v = xs[c];
    atomicAdd(ad + 4 * c + 0, v.x * w);
    atomicAdd(ad + 4 * c + 1, v.y * w);
    atomicAdd(ad + 4 * c + 2, v.z * w);
    atomicAdd(ad + 4 * c + 3, v.w * w);
  }
}

// ---------------- finish: out = act(agg + xw*dinv^2 + b) -------------------

__global__ void k_finish(const float* __restrict__ agg,
                         const float* __restrict__ xw,
                         const float* __restrict__ dinv,
                         const float* __restrict__ bias,
                         float* __restrict__ out,
                         int n, int F, int do_relu) {
  long idx = (long)blockIdx.x * blockDim.x + threadIdx.x;
  if (idx >= (long)n * F) return;
  const int i = (int)(idx / F);
  const int f = (int)(idx % F);
  const float dv = dinv[i];
  float v = agg[idx] + xw[idx] * dv * dv + bias[f];
  out[idx] = do_relu ? fmaxf(v, 0.0f) : v;
}

// ---------------------------------------------------------------------------

extern "C" void kernel_launch(void* const* d_in, const int* in_sizes, int n_in,
                              void* d_out, int out_size, void* d_ws, size_t ws_size,
                              hipStream_t stream) {
  const float*     x  = (const float*)d_in[0];
  const long long* ei = (const long long*)d_in[1];   // int64 [2, E] row-major
  const float*     W1 = (const float*)d_in[2];
  const float*     b1 = (const float*)d_in[3];
  const float*     W2 = (const float*)d_in[4];
  const float*     b2 = (const float*)d_in[5];
  const float*     W3 = (const float*)d_in[6];
  const float*     b3 = (const float*)d_in[7];

  const int NFEAT = 128, HID = 64, NCLS = 16;
  const int N = in_sizes[0] / NFEAT;     // 100000  (6250 * 16, exact)
  const int E = in_sizes[1] / 2;         // 1600000
  const long long* src = ei;
  const long long* dst = ei + E;

  // workspace layout
  float* ws   = (float*)d_ws;
  float* dinv = ws;                                  // N floats
  float* norm = dinv + N;                            // E floats
  float* bufA = norm + E;                            // N*64 floats (xw)
  float* bufB = bufA + (size_t)N * HID;              // N*64 floats (agg -> h)
  __bf16* Wt1 = (__bf16*)(bufB + (size_t)N * HID);   // 64*128 bf16
  __bf16* Wt2 = Wt1 + (size_t)HID * NFEAT;           // 64*64 bf16
  __bf16* Wt3 = Wt2 + (size_t)HID * HID;             // 16*64 bf16

  const int T = 256;
  auto cdiv = [](long a, long b) { return (int)((a + b - 1) / b); };

  // ---- normalization coefficients ----
  k_init_deg<<<cdiv(N, T), T, 0, stream>>>(dinv, N);
  k_count_deg<<<cdiv(E, T), T, 0, stream>>>(dst, dinv, E);
  k_rsqrt_inplace<<<cdiv(N, T), T, 0, stream>>>(dinv, N);
  k_edge_norm<<<cdiv(E, T), T, 0, stream>>>(src, dst, dinv, norm, E);

  // ---- weight transposes (tiny) ----
  k_weight_t_bf16<<<cdiv(NFEAT * HID, T), T, 0, stream>>>(W1, Wt1, NFEAT, HID);
  k_weight_t_bf16<<<cdiv(HID * HID, T), T, 0, stream>>>(W2, Wt2, HID, HID);
  k_weight_t_bf16<<<cdiv(HID * NCLS, T), T, 0, stream>>>(W3, Wt3, HID, NCLS);

  const int tilesM = N / 16;             // 6250

  // ---- layer 1: 128 -> 64, relu ----
  k_gemm_bf16_wmma<128, 64><<<tilesM, 32 * (64 / 16), 0, stream>>>(x, Wt1, bufA);
  k_zero<<<cdiv((long)N * HID, T), T, 0, stream>>>(bufB, (long)N * HID);
  k_aggregate<<<cdiv(E, T), T, 0, stream>>>(src, dst, norm, bufA, bufB, E, HID);
  k_finish<<<cdiv((long)N * HID, T), T, 0, stream>>>(bufB, bufA, dinv, b1, bufB, N, HID, 1);

  // ---- layer 2: 64 -> 64, relu ----
  k_gemm_bf16_wmma<64, 64><<<tilesM, 32 * (64 / 16), 0, stream>>>(bufB, Wt2, bufA);
  k_zero<<<cdiv((long)N * HID, T), T, 0, stream>>>(bufB, (long)N * HID);
  k_aggregate<<<cdiv(E, T), T, 0, stream>>>(src, dst, norm, bufA, bufB, E, HID);
  k_finish<<<cdiv((long)N * HID, T), T, 0, stream>>>(bufB, bufA, dinv, b2, bufB, N, HID, 1);

  // ---- layer 3: 64 -> 16, no relu, to d_out ----
  k_gemm_bf16_wmma<64, 16><<<tilesM, 32 * (16 / 16), 0, stream>>>(bufB, Wt3, bufA);
  k_zero<<<cdiv((long)N * NCLS, T), T, 0, stream>>>(bufB, (long)N * NCLS);
  k_aggregate<<<cdiv(E, T), T, 0, stream>>>(src, dst, norm, bufA, bufB, E, NCLS);
  k_finish<<<cdiv((long)N * NCLS, T), T, 0, stream>>>(bufB, bufA, dinv, b3,
                                                     (float*)d_out, N, NCLS, 0);
}
